// SFMStage_10977936408911
// MI455X (gfx1250) — matrix operand order repaired and checked by reading, not compiled, not run on hardware
//
#include <hip/hip_runtime.h>

typedef __attribute__((ext_vector_type(16))) _Float16 v16h;
typedef __attribute__((ext_vector_type(8)))  float    v8f;

#define NROI 128
#define CCH  256
#define MHH  28
#define MWW  28
#define HW   784
#define KCLS 80
#define PPTS 196
#define PPAD 208
#define SHH  200
#define SWW  200
#define SPIX 40000
#define FCIN 416

// ---------------- workspace layout (bytes) ----------------
#define OFF_SEM   ((size_t)0)            // sem fp32 (row-major c x 40000)      40,960,000
#define OFF_INST  ((size_t)40960000)     // inst_log fp32                       32,112,640
#define OFF_DET   ((size_t)73072640)     // det_log fp32                       32,112,640
#define OFF_FUSED ((size_t)0)            // fused fp32, overlays dead sem/inst/det: 102,760,448
#define OFF_IDX   ((size_t)105185280)    // point indices                          100,352
#define OFF_XA    ((size_t)105285632)    // Xa f16 [208][416] per ROI           22,151,168
#define OFF_SEMT  ((size_t)105285632)    // semT f16 [40000][256], dead before Xa: 20,480,000
#define OFF_XB    ((size_t)127436800)    // Xb f16                              22,151,168
#define OFF_PTS   ((size_t)149587968)    // pts fp32 (n,c,p)                    25,690,112
#define OFF_REF   ((size_t)175278080)    // featsT/refined f16 [n][784][256]    51,380,224
#define OFF_W16   ((size_t)226658304)    // f16 weights                          1,196,032
// f16 weight sub-offsets (elements)
#define W16_SEM   ((size_t)0)
#define W16_ID    ((size_t)65536)
#define W16_FC0   ((size_t)106496)
#define W16_FC1   ((size_t)212992)
#define W16_FC2   ((size_t)319488)
#define W16_LOG   ((size_t)425984)
#define W16_FUSE  ((size_t)532480)

// ---------------- WMMA fragment helpers (CDNA5 wave32 layouts) ----------------
// A (16x32 f16): lane l (0..15): M=l, K = {0..7, 16..23}; lanes 16..31: K = {8..15, 24..31}
static __device__ __forceinline__ void load_a_frag(const _Float16* W, int ldw, int m,
                                                   int kbase, int half, v16h& a) {
  const _Float16* r = W + (long)m * ldw + kbase + 8 * half;
#pragma unroll
  for (int e = 0; e < 8; ++e) a[e] = r[e];
#pragma unroll
  for (int e = 0; e < 8; ++e) a[8 + e] = r[16 + e];
}

static __device__ __forceinline__ v8f wmma_step(v16h a, v16h b, v8f c) {
  return __builtin_amdgcn_wmma_f32_16x16x32_f16(false, a, false, b, (short)0, c, false, false);
}

// Main loop, B stored k-contiguous (column-major, f16). Per k-tile: load the B fragment
// (2 x b128 per lane) and ALL G A-fragments up front (one load clause), then issue G
// back-to-back v_wmma — lets the scheduler use partial loadcnt waits instead of full drains.
template <int G>
static __device__ __forceinline__ void gemm_main(const _Float16* W, int ldw, int mbase,
                                                 const _Float16* B, long ldk, int nkt, int col,
                                                 int half, int l, v8f (&acc)[G]) {
#pragma unroll 1
  for (int kt = 0; kt < nkt; ++kt) {
    const _Float16* p = B + (long)col * ldk + kt * 32 + 8 * half;
    v16h bf;
#pragma unroll
    for (int e = 0; e < 8; ++e) { bf[e] = p[e]; bf[8 + e] = p[16 + e]; }
    if (kt + 1 < nkt) __builtin_prefetch((const void*)(p + 32), 0, 1);
    v16h af[G];
#pragma unroll
    for (int g = 0; g < G; ++g)
      load_a_frag(W, ldw, (mbase + g) * 16 + l, kt * 32, half, af[g]);
#pragma unroll
    for (int g = 0; g < G; ++g) acc[g] = wmma_step(af[g], bf, acc[g]);
  }
}

// ---------------- weight fp32 -> f16 ----------------
__global__ void cvt_f16_kernel(const float* __restrict__ src, _Float16* __restrict__ dst, int n) {
  int t = blockIdx.x * 256 + threadIdx.x;
  if (t < n) dst[t] = (_Float16)src[t];
}

// ---------------- LDS-tiled transpose + fp32->f16: src (rows x cols) -> dst (cols x rows) ----
// block (32,8), grid ((cols+31)/32, (rows+31)/32, batch)
__global__ void transpose_cvt_kernel(const float* __restrict__ src, _Float16* __restrict__ dst,
                                     int rows, int cols, long src_batch, long dst_batch) {
  __shared__ float tile[32][33];
  int b = blockIdx.z;
  const float* S = src + (long)b * src_batch;
  _Float16* D = dst + (long)b * dst_batch;
  int c0 = blockIdx.x * 32, r0 = blockIdx.y * 32;
  int tx = threadIdx.x, ty = threadIdx.y;
#pragma unroll
  for (int i = 0; i < 32; i += 8) {
    int r = r0 + ty + i, c = c0 + tx;
    if (r < rows && c < cols) tile[ty + i][tx] = S[(long)r * cols + c];
  }
  __syncthreads();
#pragma unroll
  for (int i = 0; i < 32; i += 8) {
    int c = c0 + ty + i, r = r0 + tx;
    if (c < cols && r < rows) D[(long)c * rows + r] = (_Float16)tile[tx][ty + i];
  }
}

// ---------------- sem = relu(Wsem @ semantic + b) : M=256, N=40000, K=256 ----------------
// grid (2500, 2), block 32; G=8; B = semT (f16, [40000][256])
__global__ void gemm_sem_kernel(const _Float16* __restrict__ W, const _Float16* __restrict__ Bt,
                                const float* __restrict__ bias, float* __restrict__ out) {
  int lane = threadIdx.x & 31, half = lane >> 4, l = lane & 15;
  int col = blockIdx.x * 16 + l;
  int mbase = blockIdx.y * 8;
  v8f acc[8];
#pragma unroll
  for (int g = 0; g < 8; ++g) acc[g] = {};
  gemm_main<8>(W, 256, mbase, Bt, 256, 8, col, half, l, acc);
#pragma unroll
  for (int g = 0; g < 8; ++g) {
#pragma unroll
    for (int v = 0; v < 8; ++v) {
      int row = (mbase + g) * 16 + v + 8 * half;
      float x = acc[g][v] + bias[row];
      out[(long)row * SPIX + col] = x > 0.f ? x : 0.f;
    }
  }
}

// ---------------- inst_log/det_log: per ROI, M=160 (80+80), N=784, K=256 ----------------
// grid (49, 1, 128), block 32; G=10; B = featsT (f16, per ROI [784][256])
__global__ void gemm_instdet_kernel(const _Float16* __restrict__ W, const _Float16* __restrict__ featsT,
                                    const float* __restrict__ b_inst, const float* __restrict__ b_det,
                                    float* __restrict__ inst_log, float* __restrict__ det_log) {
  int lane = threadIdx.x & 31, half = lane >> 4, l = lane & 15;
  int n = blockIdx.z;
  const _Float16* B = featsT + (long)n * HW * CCH;
  int col = blockIdx.x * 16 + l;
  v8f acc[10];
#pragma unroll
  for (int g = 0; g < 10; ++g) acc[g] = {};
  gemm_main<10>(W, 256, 0, B, 256, 8, col, half, l, acc);
#pragma unroll
  for (int g = 0; g < 10; ++g) {
#pragma unroll
    for (int v = 0; v < 8; ++v) {
      int row = g * 16 + v + 8 * half;
      if (row < KCLS)
        inst_log[((long)n * KCLS + row) * HW + col] = acc[g][v] + b_inst[row];
      else
        det_log[((long)n * KCLS + (row - KCLS)) * HW + col] = acc[g][v] + b_det[row - KCLS];
    }
  }
}

// ---------------- class-selected preds ----------------
__global__ void preds_kernel(const float* __restrict__ inst_log, const float* __restrict__ det_log,
                             const int* __restrict__ labels, float* __restrict__ out_inst,
                             float* __restrict__ out_det) {
  long t = (long)blockIdx.x * 256 + threadIdx.x;
  if (t >= (long)NROI * HW) return;
  int n = (int)(t / HW), p = (int)(t % HW);
  int lbl = labels[n];
  out_inst[t] = inst_log[((long)n * KCLS + lbl) * HW + p];
  out_det[t]  = det_log[((long)n * KCLS + lbl) * HW + p];
}

// ---------------- exact top-196 of 784 per ROI (bitonic sort: value desc, index asc) -------
__global__ void topk_kernel(const float* __restrict__ det_log, const int* __restrict__ labels,
                            int* __restrict__ pidx) {
  __shared__ float sv[1024];
  __shared__ int   si[1024];
  int n = blockIdx.x, tid = threadIdx.x;
  const float* src = det_log + ((long)n * KCLS + labels[n]) * HW;
  for (int i = tid; i < 1024; i += 512) {
    sv[i] = (i < HW) ? src[i] : -3.4e38f;
    si[i] = i;
  }
  __syncthreads();
  for (int k = 2; k <= 1024; k <<= 1) {
    for (int j = k >> 1; j > 0; j >>= 1) {
      for (int t = tid; t < 1024; t += 512) {
        int ixj = t ^ j;
        if (ixj > t) {
          float v0 = sv[t], v1 = sv[ixj];
          int i0 = si[t], i1 = si[ixj];
          bool prec = (v1 > v0) || (v1 == v0 && i1 < i0);  // elem[ixj] precedes elem[t] (desc)
          bool desc = ((t & k) == 0);
          if (desc == prec) { sv[t] = v1; sv[ixj] = v0; si[t] = i1; si[ixj] = i0; }
        }
      }
      __syncthreads();
    }
  }
  for (int p = tid; p < PPTS; p += 512) pidx[n * PPTS + p] = si[p];
}

// ---------------- build X = concat[fine, c_inst, c_det], column-major [PPAD][416] f16 ------
__global__ void build_x_kernel(const float* __restrict__ sem, const float* __restrict__ inst_log,
                               const float* __restrict__ det_log, const float* __restrict__ rois,
                               const int* __restrict__ pidx, _Float16* __restrict__ Xa,
                               _Float16* __restrict__ Xb) {
  int n = blockIdx.x, tid = threadIdx.x;  // 256 threads
  const float* r = rois + n * 5;
  float x1 = r[1], y1 = r[2], x2 = r[3], y2 = r[4];
  _Float16* XA = Xa + (long)n * PPAD * FCIN;
  _Float16* XB = Xb + (long)n * PPAD * FCIN;

  // rows 0..255 of each column: bilinear-sampled semantic features (thread = channel)
  int c = tid;
  const float* ch = sem + (long)c * SPIX;
  for (int p = 0; p < PPTS; ++p) {
    int pi = pidx[n * PPTS + p];
    float rel_x = ((pi % MWW) + 0.5f) / (float)MWW;
    float rel_y = ((pi / MWW) + 0.5f) / (float)MHH;
    float px = (x1 + rel_x * (x2 - x1)) * 0.25f - 0.5f;
    float py = (y1 + rel_y * (y2 - y1)) * 0.25f - 0.5f;
    float xf = floorf(px), yf = floorf(py);
    int x0 = (int)xf, y0 = (int)yf;
    float wx1 = px - xf, wy1 = py - yf;
    float w00 = (1.f - wx1) * (1.f - wy1), w10 = wx1 * (1.f - wy1);
    float w01 = (1.f - wx1) * wy1, w11 = wx1 * wy1;
    float acc = 0.f;
    if (x0 >= 0 && x0 < SWW && y0 >= 0 && y0 < SHH)                 acc += ch[y0 * SWW + x0] * w00;
    if (x0 + 1 >= 0 && x0 + 1 < SWW && y0 >= 0 && y0 < SHH)         acc += ch[y0 * SWW + x0 + 1] * w10;
    if (x0 >= 0 && x0 < SWW && y0 + 1 >= 0 && y0 + 1 < SHH)         acc += ch[(y0 + 1) * SWW + x0] * w01;
    if (x0 + 1 >= 0 && x0 + 1 < SWW && y0 + 1 >= 0 && y0 + 1 < SHH) acc += ch[(y0 + 1) * SWW + x0 + 1] * w11;
    XA[(long)p * FCIN + c] = (_Float16)acc;
  }
  // rows 256..415: gathered inst/det logits at point indices (constant across layers)
  for (int t = tid; t < 160 * PPTS; t += 256) {
    int rr = t % 160, p = t / 160;
    int pi = pidx[n * PPTS + p];
    float v = (rr < KCLS) ? inst_log[((long)n * KCLS + rr) * HW + pi]
                          : det_log[((long)n * KCLS + (rr - KCLS)) * HW + pi];
    _Float16 h = (_Float16)v;
    XA[(long)p * FCIN + CCH + rr] = h;
    XB[(long)p * FCIN + CCH + rr] = h;
  }
  // zero N-padding columns (columns are independent in all GEMMs)
  for (int t = tid; t < (PPAD - PPTS) * FCIN; t += 256) {
    int pc = PPTS + t / FCIN, row = t % FCIN;
    XA[(long)pc * FCIN + row] = (_Float16)0.f;
    XB[(long)pc * FCIN + row] = (_Float16)0.f;
  }
}

// ---------------- MLP layer: Xdst rows[0:256] = relu(W(256x416) @ Xsrc + b), f16 out -------
// grid (13, 2, 128), block 32; G=8
__global__ void gemm_fc_kernel(const _Float16* __restrict__ W, const _Float16* __restrict__ Xsrc,
                               const float* __restrict__ bias, _Float16* __restrict__ Xdst) {
  int lane = threadIdx.x & 31, half = lane >> 4, l = lane & 15;
  int n = blockIdx.z;
  const _Float16* B = Xsrc + (long)n * PPAD * FCIN;
  int col = blockIdx.x * 16 + l;
  int mbase = blockIdx.y * 8;
  v8f acc[8];
#pragma unroll
  for (int g = 0; g < 8; ++g) acc[g] = {};
  gemm_main<8>(W, FCIN, mbase, B, FCIN, 13, col, half, l, acc);
  _Float16* Y = Xdst + (long)n * PPAD * FCIN;
#pragma unroll
  for (int g = 0; g < 8; ++g) {
#pragma unroll
    for (int v = 0; v < 8; ++v) {
      int row = (mbase + g) * 16 + v + 8 * half;
      float x = acc[g][v] + bias[row];
      Y[(long)col * FCIN + row] = (_Float16)(x > 0.f ? x : 0.f);
    }
  }
}

// ---------------- logits: pts(n,256,196) fp32 = Wlog @ X + b (no relu) ----------------
__global__ void gemm_logits_kernel(const _Float16* __restrict__ W, const _Float16* __restrict__ Xsrc,
                                   const float* __restrict__ bias, float* __restrict__ pts) {
  int lane = threadIdx.x & 31, half = lane >> 4, l = lane & 15;
  int n = blockIdx.z;
  const _Float16* B = Xsrc + (long)n * PPAD * FCIN;
  int col = blockIdx.x * 16 + l;
  int mbase = blockIdx.y * 8;
  v8f acc[8];
#pragma unroll
  for (int g = 0; g < 8; ++g) acc[g] = {};
  gemm_main<8>(W, FCIN, mbase, B, FCIN, 13, col, half, l, acc);
  if (col < PPTS) {
#pragma unroll
    for (int g = 0; g < 8; ++g) {
#pragma unroll
      for (int v = 0; v < 8; ++v) {
        int row = (mbase + g) * 16 + v + 8 * half;
        pts[((long)n * CCH + row) * PPTS + col] = acc[g][v] + bias[row];
      }
    }
  }
}

// ---------------- scatter pts into refined (= featsT, [n][784][256] f16) -------------------
__global__ void scatter_pts_kernel(const float* __restrict__ pts, const int* __restrict__ pidx,
                                   _Float16* __restrict__ refined) {
  long t = (long)blockIdx.x * 256 + threadIdx.x;
  if (t >= (long)NROI * CCH * PPTS) return;
  int c = (int)(t % CCH);
  long rem = t / CCH;
  int p = (int)(rem % PPTS);
  int n = (int)(rem / PPTS);
  refined[((long)n * HW + pidx[n * PPTS + p]) * CCH + c] =
      (_Float16)pts[((long)n * CCH + c) * PPTS + p];
}

// ---------------- fused = relu(Wfuse @ refined + b) : per ROI M=256, N=784, K=256 ----------
// grid (49, 2, 128), block 32; G=8; B = refined ([784][256] f16)
__global__ void gemm_fuse_kernel(const _Float16* __restrict__ W, const _Float16* __restrict__ refined,
                                 const float* __restrict__ bias, float* __restrict__ fused) {
  int lane = threadIdx.x & 31, half = lane >> 4, l = lane & 15;
  int n = blockIdx.z;
  const _Float16* B = refined + (long)n * HW * CCH;
  int col = blockIdx.x * 16 + l;
  int mbase = blockIdx.y * 8;
  v8f acc[8];
#pragma unroll
  for (int g = 0; g < 8; ++g) acc[g] = {};
  gemm_main<8>(W, 256, mbase, B, 256, 8, col, half, l, acc);
#pragma unroll
  for (int g = 0; g < 8; ++g) {
#pragma unroll
    for (int v = 0; v < 8; ++v) {
      int row = (mbase + g) * 16 + v + 8 * half;
      float x = acc[g][v] + bias[row];
      fused[((long)n * CCH + row) * HW + col] = x > 0.f ? x : 0.f;
    }
  }
}

// ---------------- 2x bilinear upsample (align_corners=False, edge clamp) + relu -----------
__global__ void upsample_kernel(const float* __restrict__ fused, float* __restrict__ out) {
  long t = (long)blockIdx.x * 256 + threadIdx.x;
  long total = (long)NROI * CCH * 56 * 56;
  if (t >= total) return;
  int ox = (int)(t % 56);
  int oy = (int)((t / 56) % 56);
  long nc = t / (56 * 56);
  float sy = oy * 0.5f - 0.25f, sx = ox * 0.5f - 0.25f;
  float yf = floorf(sy), xf = floorf(sx);
  int y0 = (int)yf, x0 = (int)xf;
  float wy = sy - yf, wx = sx - xf;
  int y0c = y0 < 0 ? 0 : (y0 > 27 ? 27 : y0);
  int y1c = (y0 + 1) < 0 ? 0 : ((y0 + 1) > 27 ? 27 : y0 + 1);
  int x0c = x0 < 0 ? 0 : (x0 > 27 ? 27 : x0);
  int x1c = (x0 + 1) < 0 ? 0 : ((x0 + 1) > 27 ? 27 : x0 + 1);
  const float* f = fused + nc * HW;
  float v = f[y0c * 28 + x0c] * (1.f - wy) * (1.f - wx) + f[y0c * 28 + x1c] * (1.f - wy) * wx +
            f[y1c * 28 + x0c] * wy * (1.f - wx) + f[y1c * 28 + x1c] * wy * wx;
  out[t] = v > 0.f ? v : 0.f;
}

extern "C" void kernel_launch(void* const* d_in, const int* in_sizes, int n_in,
                              void* d_out, int out_size, void* d_ws, size_t ws_size,
                              hipStream_t stream) {
  const float* instance_feats = (const float*)d_in[0];
  const float* semantic_feat  = (const float*)d_in[1];
  const float* rois           = (const float*)d_in[2];
  const int*   roi_labels     = (const int*)d_in[3];
  const float* w_sem    = (const float*)d_in[5];
  const float* b_sem    = (const float*)d_in[6];
  const float* w_inst   = (const float*)d_in[7];
  const float* b_inst   = (const float*)d_in[8];
  const float* w_det    = (const float*)d_in[9];
  const float* b_det    = (const float*)d_in[10];
  const float* fc_w0    = (const float*)d_in[11];
  const float* fc_b0    = (const float*)d_in[12];
  const float* fc_w1    = (const float*)d_in[13];
  const float* fc_b1    = (const float*)d_in[14];
  const float* fc_w2    = (const float*)d_in[15];
  const float* fc_b2    = (const float*)d_in[16];
  const float* w_logits = (const float*)d_in[17];
  const float* b_logits = (const float*)d_in[18];
  const float* w_fuse   = (const float*)d_in[19];
  const float* b_fuse   = (const float*)d_in[20];

  char* ws = (char*)d_ws;
  float*    sem      = (float*)(ws + OFF_SEM);
  float*    inst_log = (float*)(ws + OFF_INST);
  float*    det_log  = (float*)(ws + OFF_DET);
  float*    fused    = (float*)(ws + OFF_FUSED);
  int*      pidx     = (int*)(ws + OFF_IDX);
  _Float16* Xa       = (_Float16*)(ws + OFF_XA);
  _Float16* semT     = (_Float16*)(ws + OFF_SEMT);   // overlays Xa slot; dead before build_x
  _Float16* Xb       = (_Float16*)(ws + OFF_XB);
  float*    pts      = (float*)(ws + OFF_PTS);
  _Float16* refined  = (_Float16*)(ws + OFF_REF);    // featsT == initial refined
  _Float16* wf16     = (_Float16*)(ws + OFF_W16);

  float* out_inst = (float*)d_out;
  float* out_det  = out_inst + NROI * HW;
  float* out_up   = out_inst + 2 * NROI * HW;

  // weights -> f16
  cvt_f16_kernel<<<(65536 + 255) / 256, 256, 0, stream>>>(w_sem, wf16 + W16_SEM, 65536);
  cvt_f16_kernel<<<(20480 + 255) / 256, 256, 0, stream>>>(w_inst, wf16 + W16_ID, 20480);
  cvt_f16_kernel<<<(20480 + 255) / 256, 256, 0, stream>>>(w_det, wf16 + W16_ID + 20480, 20480);
  cvt_f16_kernel<<<(106496 + 255) / 256, 256, 0, stream>>>(fc_w0, wf16 + W16_FC0, 106496);
  cvt_f16_kernel<<<(106496 + 255) / 256, 256, 0, stream>>>(fc_w1, wf16 + W16_FC1, 106496);
  cvt_f16_kernel<<<(106496 + 255) / 256, 256, 0, stream>>>(fc_w2, wf16 + W16_FC2, 106496);
  cvt_f16_kernel<<<(106496 + 255) / 256, 256, 0, stream>>>(w_logits, wf16 + W16_LOG, 106496);
  cvt_f16_kernel<<<(65536 + 255) / 256, 256, 0, stream>>>(w_fuse, wf16 + W16_FUSE, 65536);

  // B-operand transposes to k-contiguous f16
  transpose_cvt_kernel<<<dim3(SPIX / 32, CCH / 32, 1), dim3(32, 8), 0, stream>>>(
      semantic_feat, semT, CCH, SPIX, 0, 0);
  transpose_cvt_kernel<<<dim3((HW + 31) / 32, CCH / 32, NROI), dim3(32, 8), 0, stream>>>(
      instance_feats, refined, CCH, HW, (long)CCH * HW, (long)HW * CCH);

  // semantic 1x1 conv + relu  (M=256, N=40000, K=256)
  gemm_sem_kernel<<<dim3(SPIX / 16, 2), 32, 0, stream>>>(wf16 + W16_SEM, semT, b_sem, sem);

  // inst/det logits (M=160, N=784, K=256) per ROI
  gemm_instdet_kernel<<<dim3(HW / 16, 1, NROI), 32, 0, stream>>>(wf16 + W16_ID, refined,
                                                                 b_inst, b_det, inst_log, det_log);

  // class-selected coarse / detail preds (outputs 1 & 2)
  preds_kernel<<<(NROI * HW + 255) / 256, 256, 0, stream>>>(inst_log, det_log, roi_labels,
                                                            out_inst, out_det);

  // top-196 uncertain points per ROI
  topk_kernel<<<NROI, 512, 0, stream>>>(det_log, roi_labels, pidx);

  // build point-feature matrix X (fine + c_inst + c_det), column-major
  build_x_kernel<<<NROI, 256, 0, stream>>>(sem, inst_log, det_log, rois, pidx, Xa, Xb);

  // 3-layer point MLP (ping-pong) + logits
  gemm_fc_kernel<<<dim3(PPAD / 16, 2, NROI), 32, 0, stream>>>(wf16 + W16_FC0, Xa, fc_b0, Xb);
  gemm_fc_kernel<<<dim3(PPAD / 16, 2, NROI), 32, 0, stream>>>(wf16 + W16_FC1, Xb, fc_b1, Xa);
  gemm_fc_kernel<<<dim3(PPAD / 16, 2, NROI), 32, 0, stream>>>(wf16 + W16_FC2, Xa, fc_b2, Xb);
  gemm_logits_kernel<<<dim3(PPAD / 16, 2, NROI), 32, 0, stream>>>(wf16 + W16_LOG, Xb, b_logits, pts);

  // refined = featsT with point rows replaced by pts (in place)
  scatter_pts_kernel<<<(NROI * CCH * PPTS + 255) / 256, 256, 0, stream>>>(pts, pidx, refined);

  // fuse conv + relu (fused overlays dead sem/inst/det region)
  gemm_fuse_kernel<<<dim3(HW / 16, 2, NROI), 32, 0, stream>>>(wf16 + W16_FUSE, refined, b_fuse, fused);

  // 2x bilinear upsample + relu (output 3)
  long up_total = (long)NROI * CCH * 56 * 56;
  upsample_kernel<<<(unsigned)((up_total + 255) / 256), 256, 0, stream>>>(fused, out_up);
}
